// GraphReasoner_37864431681776
// MI455X (gfx1250) — matrix operand order repaired and checked by reading, not compiled, not run on hardware
//
#include <hip/hip_runtime.h>
#include <math.h>

// ---------------------------------------------------------------------------
// GraphReasoner for MI455X (gfx1250).  Heavy GEMMs use v_wmma_f32_16x16x32_bf16.
// Key algebraic optimizations (see analysis): relation GEMM factored through
// P/Q (16x FLOP cut), edge-feature attention collapsed to an 8x8 table,
// static edge structure -> per-node 16-edge softmax (no segment atomics).
// ---------------------------------------------------------------------------

typedef __attribute__((ext_vector_type(16))) __bf16 v16bf;
typedef __attribute__((ext_vector_type(8)))  float  v8f;

#define SEQ      4096
#define H_LM     1024
#define GNN_H    256
#define HEADS    8
#define N_NODES  1024
#define N_EDGES  16384

__device__ __forceinline__ unsigned short f2bf_u(float x) {
  union { float f; unsigned u; } a; a.f = x;
  unsigned r = a.u + 0x7fffu + ((a.u >> 16) & 1u);   // round-to-nearest-even
  return (unsigned short)(r >> 16);
}
__device__ __forceinline__ float gelu_f(float x) {   // exact gelu (approximate=False)
  return 0.5f * x * (1.0f + erff(x * 0.70710678118654752f));
}
__device__ __forceinline__ float elu_f(float x) { return x > 0.f ? x : (expf(x) - 1.0f); }

// ---------------------------------------------------------------------------
// Prefix-scan: out[j] = index of j-th nonzero flag (fill 0), single block.
// ---------------------------------------------------------------------------
__global__ __launch_bounds__(1024) void k_scan(const int* __restrict__ flags, int n,
                                               int* __restrict__ out, int outn) {
  __shared__ int cnt[1024];
  int t = threadIdx.x;
  int per = n / 1024;
  int c = 0;
  for (int i = 0; i < per; ++i) c += (flags[t * per + i] != 0);
  cnt[t] = c;
  __syncthreads();
  for (int off = 1; off < 1024; off <<= 1) {          // Hillis-Steele inclusive scan
    int v = cnt[t];
    int add = (t >= off) ? cnt[t - off] : 0;
    __syncthreads();
    cnt[t] = v + add;
    __syncthreads();
  }
  int base = cnt[t] - c;                               // exclusive prefix
  for (int i = t; i < outn; i += 1024) out[i] = 0;     // fill_value = 0
  __syncthreads();
  int pos = base;
  for (int i = 0; i < per; ++i) {
    int idx = t * per + i;
    if (flags[idx] != 0) { if (pos < outn) out[pos] = idx; ++pos; }
  }
}

__global__ __launch_bounds__(256) void k_compose(const int* __restrict__ vidx,
                                                 const int* __restrict__ eidx,
                                                 int* __restrict__ entidx, int n) {
  int i = blockIdx.x * 256 + threadIdx.x;
  if (i < n) entidx[i] = vidx[eidx[i]];
}

// ---------------------------------------------------------------------------
// Generic WMMA GEMM:  C[M,N] = act( A[gather(m)] @ B + bias )
// A: f32 row-major (lda), optional row gather.  B: f32 row-major KxN (ldb).
// Block tile 128x128, K-step 32, 8 waves each computing 32x64 (2x4 wmma accs).
// A/B converted to bf16 in LDS in the exact 16x16x32 fragment layout.
// ---------------------------------------------------------------------------
#define GBM 128
#define GBN 128
#define GBK 32
#define LDSD 40   // ushort stride: 80B rows keep every 16B fragment chunk aligned

__global__ __launch_bounds__(256) void k_gemm(
    const float* __restrict__ A, int lda, const int* __restrict__ rowidx,
    const float* __restrict__ B, int ldb,
    const float* __restrict__ bias, float* __restrict__ C, int ldc,
    int M, int N, int K, int act /*0 none, 1 gelu*/) {
  __shared__ __align__(16) unsigned short As[GBM * LDSD];   // [m][k]
  __shared__ __align__(16) unsigned short Bs[GBN * LDSD];   // [n][k] (transposed)

  const int tid  = threadIdx.x;
  const int lane = tid & 31;
  const int wave = tid >> 5;
  const int wm = wave & 3;            // 4 wave-rows * 32
  const int wn = wave >> 2;           // 2 wave-cols * 64
  const int bm0 = blockIdx.y * GBM;
  const int bn0 = blockIdx.x * GBN;

  v8f acc[2][4] = {};

  // A staging: 2 threads per row, 16 consecutive K each
  const int arow = tid >> 1;
  const int akb  = (tid & 1) * 16;
  int grow = bm0 + arow;
  if (rowidx) grow = rowidx[grow];
  const float* Aptr = A + (size_t)grow * lda + akb;

  const int bnl = tid & 127;          // B staging: transpose f32[k][n] -> bf16 LDS[n][k]
  const int bkb = tid >> 7;           // 0..1

  for (int k0 = 0; k0 < K; k0 += GBK) {
    {
      const float* p = Aptr + k0;
      __builtin_prefetch(p + GBK, 0, 1);               // global_prefetch_b8 next tile
      unsigned short* dst = &As[arow * LDSD + akb];
#pragma unroll
      for (int i = 0; i < 16; ++i) dst[i] = f2bf_u(p[i]);
    }
    {
#pragma unroll
      for (int p = 0; p < 16; ++p) {
        int k = p * 2 + bkb;
        float v = B[(size_t)(k0 + k) * ldb + bn0 + bnl];
        Bs[bnl * LDSD + k] = f2bf_u(v);
      }
    }
    __syncthreads();

    union Frag { uint4 q[2]; v16bf v; };
    Frag af[2], bf[4];
    const int mr = lane & 15;
    const int hi = lane >> 4;
#pragma unroll
    for (int mt = 0; mt < 2; ++mt) {                   // A 16x32: V0-3 K{hi*8..}, V4-7 K{16+hi*8..}
      const unsigned short* base = &As[(wm * 32 + mt * 16 + mr) * LDSD];
      af[mt].q[0] = *(const uint4*)(base + hi * 8);
      af[mt].q[1] = *(const uint4*)(base + 16 + hi * 8);
    }
#pragma unroll
    for (int nt = 0; nt < 4; ++nt) {                   // B 32x16: V0-3 K{hi*8..}, V4-7 K{16+hi*8..}
      const unsigned short* base = &Bs[(wn * 64 + nt * 16 + mr) * LDSD];
      bf[nt].q[0] = *(const uint4*)(base + hi * 8);
      bf[nt].q[1] = *(const uint4*)(base + 16 + hi * 8);
    }
#pragma unroll
    for (int mt = 0; mt < 2; ++mt)
#pragma unroll
      for (int nt = 0; nt < 4; ++nt)
        acc[mt][nt] = __builtin_amdgcn_wmma_f32_16x16x32_bf16(
            false, af[mt].v, false, bf[nt].v, (short)0, acc[mt][nt], false, false);
    __syncthreads();
  }

#pragma unroll
  for (int mt = 0; mt < 2; ++mt) {
#pragma unroll
    for (int nt = 0; nt < 4; ++nt) {
      int n  = bn0 + wn * 64 + nt * 16 + (lane & 15);
      int mb = bm0 + wm * 32 + mt * 16 + (lane >> 4) * 8;
      float bv = bias ? bias[n] : 0.f;
#pragma unroll
      for (int r = 0; r < 8; ++r) {                    // C/D: VGPR r -> M = r (+8 for hi lanes)
        float v = acc[mt][nt][r] + bv;
        if (act == 1) v = gelu_f(v);
        C[(size_t)(mb + r) * ldc + n] = v;
      }
    }
  }
}

// ---------------------------------------------------------------------------
// Entity head: softmax(es)[:,1] > 0.5  <=>  es1 - es0 > 0
// ---------------------------------------------------------------------------
__global__ __launch_bounds__(256) void k_entity_score(const float* __restrict__ T1,
                                                      const float* __restrict__ eW2,
                                                      const float* __restrict__ eb2,
                                                      int* __restrict__ flags) {
  int row  = blockIdx.x * 8 + (threadIdx.x >> 5);
  int lane = threadIdx.x & 31;
  float acc = 0.f;
  for (int k = lane; k < H_LM; k += 32)
    acc += T1[(size_t)row * H_LM + k] * (eW2[k * 2 + 1] - eW2[k * 2 + 0]);
#pragma unroll
  for (int off = 16; off; off >>= 1) acc += __shfl_xor(acc, off, 32);
  if (lane == 0) flags[row] = (acc + eb2[1] - eb2[0]) > 0.f ? 1 : 0;
}

// ---------------------------------------------------------------------------
// Edge typing: rtype[e] = argmax( gelu(P[s]+Q[t]) @ rW2 + rb2 )   (P carries rb1)
// Static edge map: s = e/1023, r = e%1023, t = r + (r>=s).
// ---------------------------------------------------------------------------
__global__ __launch_bounds__(256) void k_edge_type(const float* __restrict__ P,
                                                   const float* __restrict__ Q,
                                                   const float* __restrict__ rW2,
                                                   const float* __restrict__ rb2,
                                                   int* __restrict__ rtype) {
  int e = blockIdx.x * 8 + (threadIdx.x >> 5);
  int lane = threadIdx.x & 31;
  int s = e / 1023, r = e % 1023;
  int t = r + (r >= s ? 1 : 0);
  const float4* w4 = (const float4*)rW2;               // (1024,4) row-major
  float a0 = 0, a1 = 0, a2 = 0, a3 = 0;
  for (int k = lane; k < H_LM; k += 32) {
    float v = gelu_f(P[s * H_LM + k] + Q[t * H_LM + k]);
    float4 w = w4[k];
    a0 += v * w.x; a1 += v * w.y; a2 += v * w.z; a3 += v * w.w;
  }
#pragma unroll
  for (int off = 16; off; off >>= 1) {
    a0 += __shfl_xor(a0, off, 32); a1 += __shfl_xor(a1, off, 32);
    a2 += __shfl_xor(a2, off, 32); a3 += __shfl_xor(a3, off, 32);
  }
  if (lane == 0) {
    float sc[4] = { a0 + rb2[0], a1 + rb2[1], a2 + rb2[2], a3 + rb2[3] };
    int best = 0;
    for (int j = 1; j < 4; ++j) if (sc[j] > sc[best]) best = j;   // first-max ties
    rtype[e] = best;
  }
}

// ---------------------------------------------------------------------------
// Per-(node,head) attention scores: sa = <xl, a_s>, sd = <xl, a_d>
// ---------------------------------------------------------------------------
__global__ __launch_bounds__(256) void k_node_scores(const float* __restrict__ XL, int ldx,
                                                     const float* __restrict__ a_s,
                                                     const float* __restrict__ a_d, int C,
                                                     float* __restrict__ sa,
                                                     float* __restrict__ sd) {
  int n = blockIdx.x, h = threadIdx.x >> 5, lane = threadIdx.x & 31;
  float ss = 0.f, ds = 0.f;
  for (int c = lane; c < C; c += 32) {
    float x = XL[(size_t)n * ldx + h * C + c];
    ss += x * a_s[h * C + c];
    ds += x * a_d[h * C + c];
  }
#pragma unroll
  for (int off = 16; off; off >>= 1) { ss += __shfl_xor(ss, off, 32); ds += __shfl_xor(ds, off, 32); }
  if (lane == 0) { sa[n * HEADS + h] = ss; sd[n * HEADS + h] = ds; }
}

// ---------------------------------------------------------------------------
// Edge-attention table: Te[type][h] = sum_c (edge_emb[type] @ We)[h,c] * a_e[h,c]
// (eattr has only 8 distinct rows -> entire el GEMM collapses to this 8x8 table)
// ---------------------------------------------------------------------------
__global__ __launch_bounds__(256) void k_edge_table(const float* __restrict__ edge_emb,
                                                    const float* __restrict__ We,
                                                    const float* __restrict__ a_e, int C,
                                                    float* __restrict__ Te) {
  int type = blockIdx.x >> 3, h = blockIdx.x & 7, tid = threadIdx.x;
  __shared__ float red[256];
  float p = 0.f;
  for (int c = tid; c < C; c += 256) {
    float ew = 0.f;
#pragma unroll 8
    for (int d = 0; d < 64; ++d) ew += edge_emb[type * 64 + d] * We[(size_t)d * (HEADS * C) + h * C + c];
    p += ew * a_e[h * C + c];
  }
  red[tid] = p; __syncthreads();
  for (int s = 128; s; s >>= 1) { if (tid < s) red[tid] += red[tid + s]; __syncthreads(); }
  if (tid == 0) Te[type * HEADS + h] = red[0];
}

// ---------------------------------------------------------------------------
// GAT aggregate: every dst node has exactly 16 static in-edges.
// One block per node; one wave per head: 16-way softmax + weighted row sum.
// ---------------------------------------------------------------------------
__global__ __launch_bounds__(256) void k_gat_agg(const float* __restrict__ XL, int ldx,
                                                 const float* __restrict__ sa,
                                                 const float* __restrict__ sd,
                                                 const float* __restrict__ Te,
                                                 const int* __restrict__ rtype,
                                                 const float* __restrict__ bvec, int C,
                                                 float* __restrict__ xout, int last) {
  int n = blockIdx.x, h = threadIdx.x >> 5, lane = threadIdx.x & 31;
  __shared__ float wl[HEADS][16];
  __shared__ int   sl[16];
  __shared__ float accs[1024];

  float logit = -INFINITY;
  int si = 0;
  if (lane < 16) {
    int i = lane;
    si = (n < 16 && i >= n) ? i + 1 : i;               // s in {0..16}\{n}, 16 values
    int e = si * 1023 + n - (n > si ? 1 : 0);
    float lg = sa[si * HEADS + h] + sd[n * HEADS + h] + Te[rtype[e] * HEADS + h];
    logit = lg > 0.f ? lg : 0.2f * lg;                 // leaky_relu(0.2)
  }
  float m = logit;
#pragma unroll
  for (int off = 16; off; off >>= 1) m = fmaxf(m, __shfl_xor(m, off, 32));
  float ex = (lane < 16) ? expf(logit - m) : 0.f;
  float ssum = ex;
#pragma unroll
  for (int off = 16; off; off >>= 1) ssum += __shfl_xor(ssum, off, 32);
  float wgt = ex / (ssum + 1e-16f);
  if (lane < 16) { wl[h][lane] = wgt; if (h == 0) sl[lane] = si; }
  if (last) { for (int i = threadIdx.x; i < C; i += 256) accs[i] = 0.f; }
  __syncthreads();

  if (!last) {
    for (int c = lane; c < C; c += 32) {
      float acc = 0.f;
#pragma unroll
      for (int i = 0; i < 16; ++i) acc += wl[h][i] * XL[(size_t)sl[i] * ldx + h * C + c];
      xout[(size_t)n * (HEADS * C) + h * C + c] = elu_f(acc + bvec[h * C + c]);
    }
  } else {
    for (int c = lane; c < C; c += 32) {
      float acc = 0.f;
#pragma unroll
      for (int i = 0; i < 16; ++i) acc += wl[h][i] * XL[(size_t)sl[i] * ldx + h * C + c];
      atomicAdd(&accs[c], acc * 0.125f);               // head mean
    }
    __syncthreads();
    for (int c = threadIdx.x; c < C; c += 256) xout[(size_t)n * C + c] = accs[c] + bvec[c];
  }
}

// ---------------------------------------------------------------------------
// Readout: column mean, GEMV through oW, then fused residual + LayerNorm.
// ---------------------------------------------------------------------------
__global__ __launch_bounds__(256) void k_colmean(const float* __restrict__ X, float* __restrict__ xm) {
  int c = blockIdx.x * 256 + threadIdx.x;
  float s = 0.f;
  for (int n = 0; n < N_NODES; ++n) s += X[(size_t)n * H_LM + c];
  xm[c] = s * (1.0f / N_NODES);
}

__global__ __launch_bounds__(256) void k_gemv(const float* __restrict__ xm,
                                              const float* __restrict__ oW,
                                              const float* __restrict__ ob,
                                              float* __restrict__ g) {
  int j = blockIdx.x * 256 + threadIdx.x;
  float acc = ob[j];
  for (int c = 0; c < H_LM; ++c) acc += xm[c] * oW[(size_t)c * H_LM + j];
  g[j] = acc;
}

__global__ __launch_bounds__(256) void k_residual_ln(const float* __restrict__ hs,
                                                     const float* __restrict__ g,
                                                     const float* __restrict__ lng,
                                                     const float* __restrict__ lnb,
                                                     float* __restrict__ out) {
  int row = blockIdx.x, tid = threadIdx.x;
  __shared__ float red[256];
  float v[4]; float s = 0.f;
#pragma unroll
  for (int k = 0; k < 4; ++k) {
    int c = k * 256 + tid;
    v[k] = hs[(size_t)row * H_LM + c] + g[c];
    s += v[k];
  }
  red[tid] = s; __syncthreads();
  for (int st = 128; st; st >>= 1) { if (tid < st) red[tid] += red[tid + st]; __syncthreads(); }
  float mu = red[0] * (1.0f / H_LM);
  __syncthreads();
  float s2 = 0.f;
#pragma unroll
  for (int k = 0; k < 4; ++k) { float d = v[k] - mu; s2 += d * d; }
  red[tid] = s2; __syncthreads();
  for (int st = 128; st; st >>= 1) { if (tid < st) red[tid] += red[tid + st]; __syncthreads(); }
  float rstd = rsqrtf(red[0] * (1.0f / H_LM) + 1e-5f);
#pragma unroll
  for (int k = 0; k < 4; ++k) {
    int c = k * 256 + tid;
    out[(size_t)row * H_LM + c] = (v[k] - mu) * rstd * lng[c] + lnb[c];
  }
}

// ---------------------------------------------------------------------------
// Launch sequence
// ---------------------------------------------------------------------------
extern "C" void kernel_launch(void* const* d_in, const int* in_sizes, int n_in,
                              void* d_out, int out_size, void* d_ws, size_t ws_size,
                              hipStream_t stream) {
  const float* hs     = (const float*)d_in[0];
  const int*   amask  = (const int*)  d_in[1];
  const float* node_W = (const float*)d_in[2];
  const float* node_b = (const float*)d_in[3];
  const float* eW1    = (const float*)d_in[4];
  const float* eb1    = (const float*)d_in[5];
  const float* eW2    = (const float*)d_in[6];
  const float* eb2    = (const float*)d_in[7];
  const float* rW1    = (const float*)d_in[8];
  const float* rb1    = (const float*)d_in[9];
  const float* rW2    = (const float*)d_in[10];
  const float* rb2    = (const float*)d_in[11];
  const float* eemb   = (const float*)d_in[12];
  // gnn layers flattened: (W,b,a_s,a_d,a_e,We) x 3, starting at 13
  const float* gW[3], *gb[3], *gas[3], *gad[3], *gae[3], *gWe[3];
  for (int l = 0; l < 3; ++l) {
    gW[l]  = (const float*)d_in[13 + 6 * l + 0];
    gb[l]  = (const float*)d_in[13 + 6 * l + 1];
    gas[l] = (const float*)d_in[13 + 6 * l + 2];
    gad[l] = (const float*)d_in[13 + 6 * l + 3];
    gae[l] = (const float*)d_in[13 + 6 * l + 4];
    gWe[l] = (const float*)d_in[13 + 6 * l + 5];
  }
  const float* oW  = (const float*)d_in[31];
  const float* ob  = (const float*)d_in[32];
  const float* lng = (const float*)d_in[33];
  const float* lnb = (const float*)d_in[34];

  char* ws = (char*)d_ws;
  // workspace layout (overlays: BIG holds T1 first, XL afterwards)
  float* BIG  = (float*)(ws + 0);                       // 32 MB: T1 (16 MB) then XL (<=32 MB)
  float* XA   = (float*)(ws + (33554432));              // 8 MB  x ping
  float* XB   = (float*)(ws + (33554432 + 8388608));    // 8 MB  x pong
  float* P    = (float*)(ws + (33554432 + 16777216));   // 4 MB
  float* Q    = (float*)(ws + (33554432 + 20971520));   // 4 MB
  float* XF   = (float*)(ws + (33554432 + 25165824));   // 4 MB final node features
  char*  SM   = ws + (33554432 + 29360128);
  int*   vidx   = (int*)(SM + 0);
  int*   flags  = (int*)(SM + 16384);
  int*   eidx   = (int*)(SM + 32768);
  int*   entidx = (int*)(SM + 36864);
  int*   rtype  = (int*)(SM + 40960);
  float* sa     = (float*)(SM + 106496);
  float* sd     = (float*)(SM + 139264);
  float* Te     = (float*)(SM + 172032);
  float* xmean  = (float*)(SM + 172544);
  float* gvec   = (float*)(SM + 176640);

  float* T1 = BIG;   // 4096x1024
  float* XL = BIG;   // up to 1024x8192 (T1 dead by then)

  // 1) visible-token scan (batch 0 mask)
  hipLaunchKernelGGL(k_scan, dim3(1), dim3(1024), 0, stream, amask, SEQ, vidx, SEQ);
  // 2) T1 = gelu(vh @ eW1 + eb1)   [4096x1024x1024, WMMA]
  hipLaunchKernelGGL(k_gemm, dim3(H_LM / GBN, SEQ / GBM), dim3(256), 0, stream,
                     hs, H_LM, vidx, eW1, H_LM, eb1, T1, H_LM, SEQ, H_LM, H_LM, 1);
  // 3) entity flags, 4) entity index scan, 5) compose gather indices
  hipLaunchKernelGGL(k_entity_score, dim3(SEQ / 8), dim3(256), 0, stream, T1, eW2, eb2, flags);
  hipLaunchKernelGGL(k_scan, dim3(1), dim3(1024), 0, stream, flags, SEQ, eidx, N_NODES);
  hipLaunchKernelGGL(k_compose, dim3(4), dim3(256), 0, stream, vidx, eidx, entidx, N_NODES);
  // 6) x0 = ent @ node_W + node_b   (ent gathered from hs via entidx)
  hipLaunchKernelGGL(k_gemm, dim3(GNN_H / GBN, N_NODES / GBM), dim3(256), 0, stream,
                     hs, H_LM, entidx, node_W, GNN_H, node_b, XA, GNN_H, N_NODES, GNN_H, H_LM, 0);
  // 7) P = ent @ rW1_top + rb1 ; Q = ent @ rW1_bot   (relation GEMM factorization)
  hipLaunchKernelGGL(k_gemm, dim3(H_LM / GBN, N_NODES / GBM), dim3(256), 0, stream,
                     hs, H_LM, entidx, rW1, H_LM, rb1, P, H_LM, N_NODES, H_LM, H_LM, 0);
  hipLaunchKernelGGL(k_gemm, dim3(H_LM / GBN, N_NODES / GBM), dim3(256), 0, stream,
                     hs, H_LM, entidx, rW1 + (size_t)H_LM * H_LM, H_LM, nullptr, Q, H_LM,
                     N_NODES, H_LM, H_LM, 0);
  // 8) edge types
  hipLaunchKernelGGL(k_edge_type, dim3(N_EDGES / 8), dim3(256), 0, stream, P, Q, rW2, rb2, rtype);

  // 9) GAT layers
  const int Cl[3]   = { GNN_H, GNN_H, H_LM };          // per-head width
  const int Kin[3]  = { GNN_H, HEADS * GNN_H, HEADS * GNN_H };
  const float* xin = XA;
  float* xout_buf[3] = { XB, XA, XF };
  for (int l = 0; l < 3; ++l) {
    int C = Cl[l], K = Kin[l], Nw = HEADS * C, last = (l == 2);
    hipLaunchKernelGGL(k_gemm, dim3(Nw / GBN, N_NODES / GBM), dim3(256), 0, stream,
                       xin, K, nullptr, gW[l], Nw, nullptr, XL, Nw, N_NODES, Nw, K, 0);
    hipLaunchKernelGGL(k_node_scores, dim3(N_NODES), dim3(256), 0, stream,
                       XL, Nw, gas[l], gad[l], C, sa, sd);
    hipLaunchKernelGGL(k_edge_table, dim3(64), dim3(256), 0, stream, eemb, gWe[l], gae[l], C, Te);
    hipLaunchKernelGGL(k_gat_agg, dim3(N_NODES), dim3(256), 0, stream,
                       XL, Nw, sa, sd, Te, rtype, gb[l], C, xout_buf[l], last);
    xin = xout_buf[l];
  }

  // 10) readout + residual LayerNorm over both batches
  hipLaunchKernelGGL(k_colmean, dim3(H_LM / 256), dim3(256), 0, stream, XF, xmean);
  hipLaunchKernelGGL(k_gemv, dim3(H_LM / 256), dim3(256), 0, stream, xmean, oW, ob, gvec);
  hipLaunchKernelGGL(k_residual_ln, dim3(2 * SEQ), dim3(256), 0, stream,
                     hs, gvec, lng, lnb, (float*)d_out);
}
